// CellDecoder_2688649527868
// MI455X (gfx1250) — matrix operand order repaired and checked by reading, not compiled, not run on hardware
//
#include <hip/hip_runtime.h>
#include <math.h>

// Problem sizes from the reference
constexpr int B_ = 64, T_ = 34, S_ = 35, V_ = 32000, E_ = 256, U_ = 1024;

typedef __attribute__((ext_vector_type(16))) __bf16 v16bf;
typedef __attribute__((ext_vector_type(8)))  __bf16 v8bf;
typedef __attribute__((ext_vector_type(8)))  float  v8f;
typedef __attribute__((ext_vector_type(4)))  unsigned int u32x4;
typedef __attribute__((ext_vector_type(8)))  int     i32x8;
typedef __attribute__((ext_vector_type(4)))  int     i32x4;

#define DEV static __device__ __forceinline__

// ---- WMMA fragment helpers (K-contiguous lane halves, wave32) -------------------
// A 16x32 bf16: lanes 0-15 -> M=lane, K=0..15; lanes 16-31 -> M=lane-16,
// K=16..31. rowk0 = &A[row][k0], 32B-aligned rows.
DEV v16bf load_a_bf(const __bf16* rowk0, int lane) {
  const int kb = (lane & 16) ? 16 : 0;
  v8bf lo = *(const v8bf*)(rowk0 + kb);
  v8bf hi = *(const v8bf*)(rowk0 + kb + 8);
  return __builtin_shufflevector(lo, hi, 0, 1, 2, 3, 4, 5, 6, 7,
                                 8, 9, 10, 11, 12, 13, 14, 15);
}

// B 32x16 bf16 fragment from a K-contiguous (transposed, [N][K]) layout:
// lanes 0-15 -> N=lane, K=0..15; lanes 16-31 -> N=lane-16, K=16..31.
// p = &Bt[n][k0 + ((lane&16)?16:0)] -> 16 contiguous bf16 = 2 x b128.
DEV v16bf load_bt16(const __bf16* p) {
  v8bf lo = *(const v8bf*)p;
  v8bf hi = *(const v8bf*)(p + 8);
  return __builtin_shufflevector(lo, hi, 0, 1, 2, 3, 4, 5, 6, 7,
                                 8, 9, 10, 11, 12, 13, 14, 15);
}

DEV v8f wmma_bf16(v16bf a, v16bf b, v8f c) {
  return __builtin_amdgcn_wmma_f32_16x16x32_bf16(false, a, false, b,
                                                 (short)0, c, false, false);
}

// ---- Tensor Data Mover: 2D tile load Global->LDS (D# per ISA ch.8) --------
// Loads tile_d1 rows x tile_d0 elements of 2-byte data from a row-major
// [tensor_d1, tensor_d0] tensor (row stride = stride0 elements) into LDS.
// amdgpu-toolchain (clang-23) 6-arg builtin form.
DEV void tdm_load_2d_bf16(unsigned lds_byte_off, const void* gaddr,
                          unsigned tile_d0, unsigned tile_d1,
                          unsigned tensor_d0, unsigned tensor_d1,
                          unsigned stride0) {
  unsigned long long ga = (unsigned long long)gaddr;
  u32x4 g0;
  g0[0] = 1u;                                   // count=1 (valid), user mode
  g0[1] = lds_byte_off;                         // lds_addr
  g0[2] = (unsigned)ga;                         // global_addr[31:0]
  g0[3] = (unsigned)(ga >> 32) | (2u << 30);    // global_addr[56:32] | type=2
  i32x8 g1;
  g1[0] = (int)(1u << 16);                      // wg_mask=0, data_size=1 (2B)
  g1[1] = (int)((tensor_d0 & 0xFFFFu) << 16);   // [63:48] = tensor_dim0 lo16
  g1[2] = (int)((tensor_d0 >> 16) | ((tensor_d1 & 0xFFFFu) << 16));
  g1[3] = (int)((tensor_d1 >> 16) | (tile_d0 << 16));  // dim1 hi | tile_dim0
  g1[4] = (int)tile_d1;                         // tile_dim1, tile_dim2=0
  g1[5] = (int)stride0;                         // tensor_dim0_stride[31:0]
  g1[6] = 0;                                    // stride hi, dim1_stride lo
  g1[7] = 0;
  i32x4 z4 = {0, 0, 0, 0};                      // groups 2/3 unused (2D)
  i32x8 z8 = {0, 0, 0, 0, 0, 0, 0, 0};          // extra group (clang-23 form)
  __builtin_amdgcn_tensor_load_to_lds(g0, g1, z4, z4, z8, 0);
}

// ---- weight convert+transpose: Wt[n][k] = (bf16)W[k][n] -------------------
__global__ void k_cvt_t(const float* __restrict__ W, __bf16* __restrict__ Wt,
                        int K, int N) {
  int idx = blockIdx.x * blockDim.x + threadIdx.x;
  if (idx >= K * N) return;
  int n = idx % N, k = idx / N;                 // coalesced reads of W
  Wt[(size_t)n * K + k] = (__bf16)W[(size_t)k * N + n];
}

__global__ void k_cvt(const float* __restrict__ in, __bf16* __restrict__ out,
                      int n) {
  int i = blockIdx.x * blockDim.x + threadIdx.x;
  if (i < n) out[i] = (__bf16)in[i];
}

// ---- embedding gather into bf16 [T][B][E] ---------------------------------
__global__ void k_embed(const int* __restrict__ inputs,
                        const float* __restrict__ table,
                        __bf16* __restrict__ embT) {
  int idx = blockIdx.x * blockDim.x + threadIdx.x;
  if (idx >= T_ * B_ * E_) return;
  int e = idx % E_;
  int tb = idx / E_;
  int b = tb % B_, t = tb / B_;
  int tok = inputs[b * T_ + t];
  embT[idx] = (__bf16)table[(size_t)tok * E_ + e];
}

// ---- state init -----------------------------------------------------------
__global__ void k_init(const float* __restrict__ eh, const float* __restrict__ ec,
                       float* __restrict__ h, __bf16* __restrict__ h_bf,
                       float* __restrict__ c, __bf16* __restrict__ attn_bf) {
  int i = blockIdx.x * blockDim.x + threadIdx.x;
  if (i >= B_ * U_) return;
  float hv = eh[i];
  h[i] = hv; h_bf[i] = (__bf16)hv;
  c[i] = ec[i];
  attn_bf[i] = (__bf16)0.f;
}

// ---- TDM-staged GEMM: C = A(bf16,[M,K]) x Bt(bf16,[N,K])^T (+bias) --------
// One workgroup (4 waves) owns a 64-column B panel: a single TDM DMA pulls
// [64, K] bf16 (128 KB for K=1024) into LDS, then the WG sweeps M-tiles
// reusing the panel. remap=1 maps row r=t*B+b -> out[b,t,:] (final logits).
__global__ void k_gemm_tdm(const __bf16* __restrict__ A,
                           const __bf16* __restrict__ Bt,
                           const float* __restrict__ bias, // nullable
                           float* __restrict__ Cout,
                           int M, int N, int K, int remap) {
  extern __shared__ __bf16 sB[];                // [64][K]
  const int lane = threadIdx.x & 31;
  const int wave = threadIdx.x >> 5;
  const int n0w = blockIdx.x * 64;

  if (threadIdx.x == 0) {                       // wave 0 issues the DMA
    tdm_load_2d_bf16((unsigned)(size_t)sB, Bt + (size_t)n0w * K,
                     (unsigned)K, 64u, (unsigned)K, (unsigned)N, (unsigned)K);
    __builtin_amdgcn_s_wait_tensorcnt(0);
  }
  __syncthreads();                              // publish LDS panel

  const int col = lane & 15;
  const int kbB = (lane & 16) ? 16 : 0;
  const __bf16* bcol = sB + (size_t)(wave * 16 + col) * K + kbB;
  const int n = n0w + wave * 16 + col;
  const float bb = bias ? bias[n] : 0.f;
  const int Mtiles = M / 16;

  for (int mt = blockIdx.y; mt < Mtiles; mt += gridDim.y) {
    const int m0 = mt * 16;
    const __bf16* arow = A + (size_t)(m0 + (lane & 15)) * K;
    v8f acc = {};
#pragma unroll 2
    for (int k0 = 0; k0 < K; k0 += 32) {
      acc = wmma_bf16(load_a_bf(arow + k0, lane), load_bt16(bcol + k0), acc);
    }
    const int rb = m0 + ((lane & 16) ? 8 : 0);
#pragma unroll
    for (int v = 0; v < 8; ++v) {
      int r = rb + v;
      size_t off = remap ? (((size_t)(r % B_) * T_ + (r / B_)) * (size_t)N + n)
                         : ((size_t)r * N + n);
      Cout[off] = acc[v] + bb;
    }
  }
}

// ---- LSTM gate GEMM: z[64,4096] = [x_t|attn|h](bf16) @ [Wk;Wr] + bias -----
// Segment boundaries (256, 1280) are multiples of 32 -> one segment per block.
__global__ void k_lstm_gemm(const __bf16* __restrict__ xt,
                            const __bf16* __restrict__ attn,
                            const __bf16* __restrict__ h,
                            const __bf16* __restrict__ WkT,  // [4U, E+U]
                            const __bf16* __restrict__ WrT,  // [4U, U]
                            const float* __restrict__ bias,
                            float* __restrict__ z) {         // [B, 4U]
  const int lane = threadIdx.x & 31;
  const int wave = threadIdx.x >> 5;
  const int m0 = blockIdx.x * 16;
  const int n0 = (blockIdx.y * 4 + wave) * 16;
  const int row = m0 + (lane & 15);
  const int n = n0 + (lane & 15);
  const int kbB = (lane & 16) ? 16 : 0;
  const int N = 4 * U_, K = E_ + U_ + U_;       // 4096, 2304
  v8f acc = {};
  for (int k0 = 0; k0 < K; k0 += 32) {
    const __bf16* arow;
    if (k0 < E_)           arow = xt   + (size_t)row * E_ + k0;
    else if (k0 < E_ + U_) arow = attn + (size_t)row * U_ + (k0 - E_);
    else                   arow = h    + (size_t)row * U_ + (k0 - E_ - U_);
    const __bf16* bp = (k0 < E_ + U_)
                       ? WkT + (size_t)n * (E_ + U_) + k0
                       : WrT + (size_t)n * U_ + (k0 - E_ - U_);
    __builtin_prefetch(bp + 32, 0, 1);
    acc = wmma_bf16(load_a_bf(arow, lane), load_bt16(bp + kbB), acc);
  }
  const int rb = m0 + ((lane & 16) ? 8 : 0);
  const float bb = bias[n];
#pragma unroll
  for (int v = 0; v < 8; ++v) z[(size_t)(rb + v) * N + n] = acc[v] + bb;
}

// ---- gate nonlinearity + state update (keras order i,f,g,o) ---------------
__global__ void k_gates(const float* __restrict__ z,
                        float* __restrict__ h, __bf16* __restrict__ h_bf,
                        float* __restrict__ c) {
  int idx = blockIdx.x * blockDim.x + threadIdx.x;
  if (idx >= B_ * U_) return;
  int b = idx / U_, u = idx % U_;
  const float* zr = z + (size_t)b * 4 * U_;
  float gi = zr[u], gf = zr[U_ + u], gg = zr[2 * U_ + u], go = zr[3 * U_ + u];
  float si = 1.f / (1.f + expf(-gi));
  float sf = 1.f / (1.f + expf(-gf));
  float so = 1.f / (1.f + expf(-go));
  float cn = sf * c[idx] + si * tanhf(gg);
  c[idx] = cn;
  float hn = so * tanhf(cn);
  h[idx] = hn;
  h_bf[idx] = (__bf16)hn;
}

// ---- Luong attention: score -> softmax -> context (bf16 out) --------------
__global__ void k_attn_ctx(const float* __restrict__ h,
                           const float* __restrict__ keys,   // [B,S,U] f32
                           const float* __restrict__ memory, // [B,S,U] f32
                           __bf16* __restrict__ ctx_bf) {    // [B,U]
  __shared__ float s_score[S_];
  const int b = blockIdx.x;
  const int wave = threadIdx.x >> 5, lane = threadIdx.x & 31;
  const float* hb = h + (size_t)b * U_;
  for (int s = wave; s < S_; s += 8) {
    const float* kb = keys + ((size_t)b * S_ + s) * U_;
    float acc = 0.f;
    for (int u = lane; u < U_; u += 32) acc += hb[u] * kb[u];
#pragma unroll
    for (int off = 16; off > 0; off >>= 1) acc += __shfl_xor(acc, off, 32);
    if (lane == 0) s_score[s] = acc;
  }
  __syncthreads();
  if (threadIdx.x == 0) {
    float m = s_score[0];
    for (int s = 1; s < S_; ++s) m = fmaxf(m, s_score[s]);
    float sum = 0.f;
    for (int s = 0; s < S_; ++s) { float e = expf(s_score[s] - m); s_score[s] = e; sum += e; }
    float inv = 1.f / sum;
    for (int s = 0; s < S_; ++s) s_score[s] *= inv;
  }
  __syncthreads();
  for (int u = threadIdx.x; u < U_; u += 256) {
    float acc = 0.f;
    for (int s = 0; s < S_; ++s)
      acc += s_score[s] * memory[((size_t)b * S_ + s) * U_ + u];
    ctx_bf[(size_t)b * U_ + u] = (__bf16)acc;
  }
}

// ---- attention projection: attn = [h|ctx] @ W_attn (bf16 in, bf16 out) ----
__global__ void k_attn_gemm(const __bf16* __restrict__ h,
                            const __bf16* __restrict__ ctx,
                            const __bf16* __restrict__ WaT,  // [U, 2U]
                            __bf16* __restrict__ attn_bf,    // [B, U]
                            __bf16* __restrict__ attn_t) {   // attn_all+t*B*U
  const int lane = threadIdx.x & 31;
  const int wave = threadIdx.x >> 5;
  const int m0 = blockIdx.x * 16;
  const int n0 = (blockIdx.y * 4 + wave) * 16;
  const int row = m0 + (lane & 15);
  const int n = n0 + (lane & 15);
  const int kbB = (lane & 16) ? 16 : 0;
  v8f acc = {};
  for (int k0 = 0; k0 < 2 * U_; k0 += 32) {
    const __bf16* arow = (k0 < U_) ? (h + (size_t)row * U_ + k0)
                                   : (ctx + (size_t)row * U_ + (k0 - U_));
    const __bf16* bp = WaT + (size_t)n * (2 * U_) + k0;
    acc = wmma_bf16(load_a_bf(arow, lane), load_bt16(bp + kbB), acc);
  }
  const int rb = m0 + ((lane & 16) ? 8 : 0);
#pragma unroll
  for (int v = 0; v < 8; ++v) {
    __bf16 val = (__bf16)acc[v];
    attn_bf[(size_t)(rb + v) * U_ + n] = val;
    attn_t [(size_t)(rb + v) * U_ + n] = val;
  }
}

// ---------------------------------------------------------------------------
extern "C" void kernel_launch(void* const* d_in, const int* in_sizes, int n_in,
                              void* d_out, int out_size, void* d_ws, size_t ws_size,
                              hipStream_t stream) {
  (void)in_sizes; (void)n_in; (void)out_size; (void)ws_size;
  const int*   inputs = (const int*)  d_in[0];
  const float* memory = (const float*)d_in[1];
  const float* enc_h  = (const float*)d_in[2];
  const float* enc_c  = (const float*)d_in[3];
  const float* emb    = (const float*)d_in[4];
  const float* Wk     = (const float*)d_in[5];
  const float* Wr     = (const float*)d_in[6];
  const float* lbias  = (const float*)d_in[7];
  const float* Wm     = (const float*)d_in[8];
  const float* Wa     = (const float*)d_in[9];
  const float* Wf     = (const float*)d_in[10];
  const float* bfc    = (const float*)d_in[11];
  float* out = (float*)d_out;

  // workspace carve-up (bytes, 256B-aligned chunks; ~112 MB total)
  char* wsb = (char*)d_ws;
  auto carve = [&](size_t bytes) {
    char* p = wsb;
    wsb += (bytes + 255) & ~(size_t)255;
    return (void*)p;
  };
  float*  keys     = (float*) carve((size_t)B_ * S_ * U_ * 4);
  float*  z        = (float*) carve((size_t)B_ * 4 * U_ * 4);
  float*  h        = (float*) carve((size_t)B_ * U_ * 4);
  float*  c        = (float*) carve((size_t)B_ * U_ * 4);
  __bf16* WmT      = (__bf16*)carve((size_t)U_ * U_ * 2);
  __bf16* WkT      = (__bf16*)carve((size_t)(E_ + U_) * 4 * U_ * 2);
  __bf16* WrT      = (__bf16*)carve((size_t)U_ * 4 * U_ * 2);
  __bf16* WaT      = (__bf16*)carve((size_t)2 * U_ * U_ * 2);
  __bf16* WfT      = (__bf16*)carve((size_t)U_ * V_ * 2);
  __bf16* mem_bf   = (__bf16*)carve((size_t)B_ * S_ * U_ * 2);
  __bf16* embT     = (__bf16*)carve((size_t)T_ * B_ * E_ * 2);
  __bf16* h_bf     = (__bf16*)carve((size_t)B_ * U_ * 2);
  __bf16* attn_bf  = (__bf16*)carve((size_t)B_ * U_ * 2);
  __bf16* ctx_bf   = (__bf16*)carve((size_t)B_ * U_ * 2);
  __bf16* attn_all = (__bf16*)carve((size_t)T_ * B_ * U_ * 2);

  const int TPB = 256;
  auto blocks = [](long long n) { return (unsigned)((n + 255) / 256); };

  // one-time weight cvt+transpose (all stay L2-resident afterwards)
  k_cvt_t<<<blocks((long long)U_ * U_), TPB, 0, stream>>>(Wm, WmT, U_, U_);
  k_cvt_t<<<blocks((long long)(E_ + U_) * 4 * U_), TPB, 0, stream>>>(Wk, WkT, E_ + U_, 4 * U_);
  k_cvt_t<<<blocks((long long)U_ * 4 * U_), TPB, 0, stream>>>(Wr, WrT, U_, 4 * U_);
  k_cvt_t<<<blocks((long long)2 * U_ * U_), TPB, 0, stream>>>(Wa, WaT, 2 * U_, U_);
  k_cvt_t<<<blocks((long long)U_ * V_), TPB, 0, stream>>>(Wf, WfT, U_, V_);
  k_cvt  <<<blocks((long long)B_ * S_ * U_), TPB, 0, stream>>>(memory, mem_bf, B_ * S_ * U_);

  k_embed<<<blocks((long long)T_ * B_ * E_), TPB, 0, stream>>>(inputs, emb, embT);
  k_init <<<blocks((long long)B_ * U_), TPB, 0, stream>>>(enc_h, enc_c, h, h_bf, c, attn_bf);

  const size_t panel_bytes = (size_t)64 * U_ * 2;  // 128 KB LDS B panel

  // keys = memory[B*S,U] @ W_mem  (TDM-staged, M-sweep per panel)
  k_gemm_tdm<<<dim3(U_ / 64, 35), dim3(128), panel_bytes, stream>>>(
      mem_bf, WmT, nullptr, keys, B_ * S_, U_, U_, 0);

  for (int t = 0; t < T_; ++t) {
    const __bf16* xt = embT + (size_t)t * B_ * E_;
    k_lstm_gemm<<<dim3(B_ / 16, (4 * U_ / 16) / 4), dim3(128), 0, stream>>>(
        xt, attn_bf, h_bf, WkT, WrT, lbias, z);
    k_gates<<<blocks((long long)B_ * U_), TPB, 0, stream>>>(z, h, h_bf, c);
    k_attn_ctx<<<dim3(B_), dim3(256), 0, stream>>>(h, keys, memory, ctx_bf);
    k_attn_gemm<<<dim3(B_ / 16, (U_ / 16) / 4), dim3(128), 0, stream>>>(
        h_bf, ctx_bf, WaT, attn_bf, attn_all + (size_t)t * B_ * U_);
  }

  // final logits: [T*B,U] @ W_fc + b_fc -> out[B,T,V]  (72% of all FLOPs)
  k_gemm_tdm<<<dim3(V_ / 64, 4), dim3(128), panel_bytes, stream>>>(
      attn_all, WfT, bfc, out, T_ * B_, V_, U_, 1);
}
